// Outer2Attention_11123965296646
// MI455X (gfx1250) — matrix operand order repaired and checked by reading, not compiled, not run on hardware
//
#include <hip/hip_runtime.h>
#include <hip/hip_bf16.h>
#include <math.h>

// ---------------------------------------------------------------------------
// Problem constants (reference: A=B=8192, E1=E2=C=1024)
// ---------------------------------------------------------------------------
#define NROWS 8192      // rows of seq1 / seq2
#define KDIM  1024      // E1 == E2 == C == 1024 (all GEMMs are 8192x1024x1024)

// GEMM tiling
#define BM 128          // block tile M
#define BN 128          // block tile N
#define BK 32           // K per stage (one WMMA K)
#define SKA 36          // padded LDS row stride in bf16 elems (72B: 64B data + 8B pad)

typedef __bf16 bf16_t;
typedef __attribute__((ext_vector_type(16))) __bf16 v16bf;
typedef __attribute__((ext_vector_type(8)))  __bf16 v8bf;
typedef __attribute__((ext_vector_type(4)))  __bf16 v4bf;
typedef __attribute__((ext_vector_type(8)))  float   v8f;

typedef unsigned int u32x4 __attribute__((ext_vector_type(4)));
typedef int          i32x8 __attribute__((ext_vector_type(8)));
typedef int          i32x4 __attribute__((ext_vector_type(4)));

// ---------------------------------------------------------------------------
// TDM: load a BM x BK (rows x cols) bf16 tile from a row-major [rows_total, KDIM]
// tensor into LDS, padding each 64B row with 8B so the LDS row stride is 72B
// (bank-conflict-free ds_load_b128 fragment reads).
// D# packing per CDNA5 ISA (2D tensor, groups 2/3 zero).
// Toolchain uses the 6-arg builtin: (u32x4, i32x8, i32x4, i32x4, i32x8, cpol).
// ---------------------------------------------------------------------------
__device__ __forceinline__ void tdm_load_tile(const bf16_t* g, unsigned lds_off,
                                              unsigned rows_total) {
  unsigned long long ga = (unsigned long long)(uintptr_t)g;
  u32x4 g0;
  g0[0] = 1u;                                                 // count=1 (valid user D#)
  g0[1] = lds_off;                                            // lds_addr (bytes)
  g0[2] = (unsigned)ga;                                       // global_addr[31:0]
  g0[3] = ((unsigned)(ga >> 32) & 0x1ffffffu) | (2u << 30);   // addr[56:32] | type=2
  i32x8 g1;
  g1[0] = (int)((1u << 16)      // data_size = 1 -> 2 bytes (bf16)
              | (1u << 20)      // pad_enable
              | (3u << 22)      // pad_interval = 3 -> every 16 DWORDs (64B = one row)
              | (1u << 25));    // pad_amount   = 1 -> 2 DWORDs (8B)
  g1[1] = (int)(((unsigned)KDIM & 0xffffu) << 16);            // tensor_dim0[15:0]
  g1[2] = (int)((((unsigned)KDIM >> 16) & 0xffffu)
              | ((rows_total & 0xffffu) << 16));              // dim0 hi | dim1 lo
  g1[3] = (int)(((rows_total >> 16) & 0xffffu)
              | ((unsigned)BK << 16));                        // dim1 hi | tile_dim0=32
  g1[4] = (int)BM;                                            // tile_dim1=128, tile_dim2=0
  g1[5] = (int)KDIM;                                          // tensor_dim0_stride lo
  g1[6] = 0;                                                  // stride hi, dim1_stride lo
  g1[7] = 0;
  i32x4 gz4 = {0, 0, 0, 0};
  i32x8 gz8 = {0, 0, 0, 0, 0, 0, 0, 0};
  __builtin_amdgcn_tensor_load_to_lds(g0, g1, gz4, gz4, gz8, 0);
}

// ---------------------------------------------------------------------------
// WMMA fragment loaders from LDS tiles (row stride SKA elems), wave32 layouts:
// A 16x32 bf16: lane holds row (lane&15); elems 0-7 = K 8h..8h+7,
//               elems 8-15 = K 16+8h..16+8h+7   (h = lane>>4)
// B 32x16 bf16: lane holds col (lane&15); same K order as A
// ---------------------------------------------------------------------------
__device__ __forceinline__ v16bf lds_a_frag(const bf16_t* T, int m0, int lane) {
  const int r = m0 + (lane & 15);
  const int h = lane >> 4;
  const bf16_t* p = T + r * SKA + 8 * h;
  v8bf lo = *(const v8bf*)p;          // K = 8h .. 8h+7
  v8bf hi = *(const v8bf*)(p + 16);   // K = 16+8h .. 16+8h+7
  v16bf f;
#pragma unroll
  for (int i = 0; i < 8; ++i) { f[i] = lo[i]; f[i + 8] = hi[i]; }
  return f;
}

__device__ __forceinline__ v16bf lds_b_frag(const bf16_t* T, int n0, int lane) {
  const int n = n0 + (lane & 15);
  const int h = lane >> 4;
  const bf16_t* p = T + n * SKA + 8 * h;
  v8bf lo = *(const v8bf*)p;
  v8bf hi = *(const v8bf*)(p + 16);
  v16bf f;
#pragma unroll
  for (int i = 0; i < 8; ++i) { f[i] = lo[i]; f[i + 8] = hi[i]; }
  return f;
}

__device__ __forceinline__ v8f wmma_bf16(v16bf a, v16bf b, v8f c) {
  return __builtin_amdgcn_wmma_f32_16x16x32_bf16(
      /*neg_a=*/false, a, /*neg_b=*/false, b,
      /*c_mod=*/(short)0, c, /*reuse_a=*/false, /*reuse_b=*/false);
}

// ---------------------------------------------------------------------------
// GEMM:  Out[m,n] = post( A[m,:] . W[n,:] + bias[n] + ctx[n] )    (X @ W^T)
//   - TDM double-buffered LDS staging (tensor_load_to_lds + s_wait_tensorcnt)
//   - bf16 WMMA, f32 accum; optional ELU; bf16 out; optional fused column-sum
// Block = 256 thr = 8 waves; wave tile 32x64; block tile 128x128.
// Grid = (8192/128, 1024/128) = (64, 8).
// ---------------------------------------------------------------------------
__launch_bounds__(256)
__global__ void gemm_bf16_kernel(const bf16_t* __restrict__ Amat,
                                 const bf16_t* __restrict__ Wmat,
                                 const float* __restrict__ bias,
                                 const float* __restrict__ ctx,
                                 bf16_t* __restrict__ Out,
                                 float* __restrict__ colsum,
                                 int do_elu) {
  __shared__ bf16_t sA[2][BM * SKA];   // 2 x 9216 B
  __shared__ bf16_t sW[2][BN * SKA];   // 2 x 9216 B
  __shared__ float  s_col[BN];

  const int tid  = threadIdx.x;
  const int lane = tid & 31;
  const int wave = tid >> 5;
  const int bm = blockIdx.x * BM;
  const int bn = blockIdx.y * BN;
  const int wmL = (wave & 3) * 32;   // local M of wave tile (4 M-groups)
  const int wnL = (wave >> 2) * 64;  // local N of wave tile (2 N-groups)

  if (colsum && tid < BN) s_col[tid] = 0.f;  // synced by first loop barrier

  // Stage 0 DMA (wave 0 owns the TDM; TENSORcnt is per-wave)
  if (wave == 0) {
    tdm_load_tile(Amat + (size_t)bm * KDIM, (unsigned)(uintptr_t)&sA[0][0], NROWS);
    tdm_load_tile(Wmat + (size_t)bn * KDIM, (unsigned)(uintptr_t)&sW[0][0], KDIM);
  }

  v8f acc[2][4] = {};  // [mi][ni] 16x16 f32 tiles

  const int nstages = KDIM / BK;  // 32
  for (int kt = 0; kt < nstages; ++kt) {
    const int cur = kt & 1;
    if (wave == 0) __builtin_amdgcn_s_wait_tensorcnt(0);  // stage kt landed in LDS
    __syncthreads();  // RAW: data visible; WAR: buf cur^1 reads (iter kt-1) done
    if (wave == 0 && kt + 1 < nstages) {                  // prefetch stage kt+1
      const int k0 = (kt + 1) * BK;
      tdm_load_tile(Amat + (size_t)bm * KDIM + k0,
                    (unsigned)(uintptr_t)&sA[cur ^ 1][0], NROWS);
      tdm_load_tile(Wmat + (size_t)bn * KDIM + k0,
                    (unsigned)(uintptr_t)&sW[cur ^ 1][0], KDIM);
    }
    const bf16_t* At = sA[cur];
    const bf16_t* Wt = sW[cur];
    v16bf a0 = lds_a_frag(At, wmL,      lane);
    v16bf a1 = lds_a_frag(At, wmL + 16, lane);
    v16bf b0 = lds_b_frag(Wt, wnL,      lane);
    v16bf b1 = lds_b_frag(Wt, wnL + 16, lane);
    v16bf b2 = lds_b_frag(Wt, wnL + 32, lane);
    v16bf b3 = lds_b_frag(Wt, wnL + 48, lane);
    acc[0][0] = wmma_bf16(a0, b0, acc[0][0]);
    acc[0][1] = wmma_bf16(a0, b1, acc[0][1]);
    acc[0][2] = wmma_bf16(a0, b2, acc[0][2]);
    acc[0][3] = wmma_bf16(a0, b3, acc[0][3]);
    acc[1][0] = wmma_bf16(a1, b0, acc[1][0]);
    acc[1][1] = wmma_bf16(a1, b1, acc[1][1]);
    acc[1][2] = wmma_bf16(a1, b2, acc[1][2]);
    acc[1][3] = wmma_bf16(a1, b3, acc[1][3]);
  }

  // Epilogue.  C/D layout: VGPR j -> M = j (lanes 0-15) or j+8 (lanes 16-31),
  // N = lane & 15.
  const int col  = lane & 15;
  const int half = lane >> 4;
#pragma unroll
  for (int ni = 0; ni < 4; ++ni) {
    const int n = bn + wnL + ni * 16 + col;
    const float addv = (bias ? bias[n] : 0.f) + (ctx ? ctx[n] : 0.f);
    float csum = 0.f;
#pragma unroll
    for (int mi = 0; mi < 2; ++mi) {
      const int mbase = bm + wmL + mi * 16 + half * 8;
#pragma unroll
      for (int j = 0; j < 8; ++j) {
        float x = acc[mi][ni][j] + addv;
        if (do_elu) x = (x > 0.f) ? x : expm1f(x);
        csum += x;
        Out[(size_t)(mbase + j) * KDIM + n] = (bf16_t)x;
      }
    }
    if (colsum) atomicAdd(&s_col[n - bn], csum);  // ds_add_f32
  }

  if (colsum) {
    __syncthreads();
    if (tid < BN) atomicAdd(&colsum[bn + tid], s_col[tid]);
  }
}

// ---------------------------------------------------------------------------
// f32 -> bf16 conversion (vectorized x4)
// ---------------------------------------------------------------------------
__global__ void cvt_kernel(const float4* __restrict__ in, v4bf* __restrict__ out,
                           int n4) {
  const int i = blockIdx.x * 256 + threadIdx.x;
  if (i < n4) {
    float4 v = in[i];
    v4bf o;
    o[0] = (bf16_t)v.x; o[1] = (bf16_t)v.y; o[2] = (bf16_t)v.z; o[3] = (bf16_t)v.w;
    out[i] = o;
  }
}

__global__ void zero_kernel(float* __restrict__ p, int n) {
  const int i = blockIdx.x * 256 + threadIdx.x;
  if (i < n) p[i] = 0.f;
}

// ---------------------------------------------------------------------------
// ctx GEMVs:  ctx1 = W_ctx1 @ c, ctx2 = W_ctx2 @ c   (block per output row)
// ---------------------------------------------------------------------------
__global__ void ctx_gemv_kernel(const float* __restrict__ W1,
                                const float* __restrict__ W2,
                                const float* __restrict__ c,
                                float* __restrict__ ctx1,
                                float* __restrict__ ctx2) {
  const int row = blockIdx.x;  // 0..2047
  const float* Wr = (row < KDIM) ? (W1 + (size_t)row * KDIM)
                                 : (W2 + (size_t)(row - KDIM) * KDIM);
  float s = 0.f;
  for (int k = threadIdx.x; k < KDIM; k += 256) s += Wr[k] * c[k];
  __shared__ float red[8];
  const int lane = threadIdx.x & 31, wv = threadIdx.x >> 5;
  for (int o = 16; o; o >>= 1) s += __shfl_down(s, o, 32);
  if (lane == 0) red[wv] = s;
  __syncthreads();
  if (threadIdx.x == 0) {
    float t = 0.f;
    for (int i = 0; i < 8; ++i) t += red[i];
    if (row < KDIM) ctx1[row] = t; else ctx2[row - KDIM] = t;
  }
}

// ---------------------------------------------------------------------------
// logits[row] = bf16 Mtx[row,:] . f32 v[:]   (block per row)
// ---------------------------------------------------------------------------
__global__ void rowdot_kernel(const bf16_t* __restrict__ Mtx,
                              const float* __restrict__ v,
                              float* __restrict__ out) {
  const int row = blockIdx.x;
  const bf16_t* p = Mtx + (size_t)row * KDIM;
  float s = 0.f;
  for (int k = threadIdx.x; k < KDIM; k += 256) s += (float)p[k] * v[k];
  __shared__ float red[8];
  const int lane = threadIdx.x & 31, wv = threadIdx.x >> 5;
  for (int o = 16; o; o >>= 1) s += __shfl_down(s, o, 32);
  if (lane == 0) red[wv] = s;
  __syncthreads();
  if (threadIdx.x == 0) {
    float t = 0.f;
    for (int i = 0; i < 8; ++i) t += red[i];
    out[row] = t;
  }
}

// ---------------------------------------------------------------------------
// softmax over n logits -> weights (single block; n = 8192 is tiny)
// ---------------------------------------------------------------------------
__global__ void softmax_kernel(const float* __restrict__ logits,
                               float* __restrict__ w, int n) {
  __shared__ float red[8];
  __shared__ float bcast;
  const int tid = threadIdx.x, lane = tid & 31, wv = tid >> 5;

  float m = -3.402823466e38f;
  for (int i = tid; i < n; i += 256) m = fmaxf(m, logits[i]);
  for (int o = 16; o; o >>= 1) m = fmaxf(m, __shfl_down(m, o, 32));
  if (lane == 0) red[wv] = m;
  __syncthreads();
  if (tid == 0) {
    float t = red[0];
    for (int i = 1; i < 8; ++i) t = fmaxf(t, red[i]);
    bcast = t;
  }
  __syncthreads();
  const float gmax = bcast;

  float s = 0.f;
  for (int i = tid; i < n; i += 256) s += __expf(logits[i] - gmax);
  for (int o = 16; o; o >>= 1) s += __shfl_down(s, o, 32);
  __syncthreads();
  if (lane == 0) red[wv] = s;
  __syncthreads();
  if (tid == 0) {
    float t = 0.f;
    for (int i = 0; i < 8; ++i) t += red[i];
    bcast = t;
  }
  __syncthreads();
  const float inv = 1.f / bcast;
  for (int i = tid; i < n; i += 256) w[i] = __expf(logits[i] - gmax) * inv;
}

// ---------------------------------------------------------------------------
// att[e] = sum_r seq[r,e] * w[r]   (f32 inputs; grid (E/256, row-slabs))
// ---------------------------------------------------------------------------
__global__ void att_kernel(const float* __restrict__ seq,
                           const float* __restrict__ w,
                           float* __restrict__ out) {
  const int e  = blockIdx.x * 256 + threadIdx.x;  // 0..1023
  const int r0 = blockIdx.y * 256;                // 32 slabs of 256 rows
  float s = 0.f;
  for (int r = r0; r < r0 + 256; ++r) s += seq[(size_t)r * KDIM + e] * w[r];
  atomicAdd(&out[e], s);
}

// ---------------------------------------------------------------------------
extern "C" void kernel_launch(void* const* d_in, const int* in_sizes, int n_in,
                              void* d_out, int out_size, void* d_ws, size_t ws_size,
                              hipStream_t stream) {
  (void)in_sizes; (void)n_in; (void)out_size; (void)ws_size;
  const float* seq1     = (const float*)d_in[0];
  const float* seq2     = (const float*)d_in[1];
  const float* context  = (const float*)d_in[2];
  const float* Wctx1    = (const float*)d_in[3];
  const float* Wmlp1    = (const float*)d_in[4];
  const float* bmlp1    = (const float*)d_in[5];
  const float* Wctx2    = (const float*)d_in[6];
  const float* Wmlp2    = (const float*)d_in[7];
  const float* bmlp2    = (const float*)d_in[8];
  const float* Winner   = (const float*)d_in[9];
  float* out = (float*)d_out;

  char* ws = (char*)d_ws;
  size_t off = 0;
  auto take = [&](size_t bytes) -> char* {
    char* p = ws + off;
    off += (bytes + 255) & ~(size_t)255;
    return p;
  };
  const size_t seq_bytes = (size_t)NROWS * KDIM * sizeof(bf16_t);
  const size_t w_bytes   = (size_t)KDIM * KDIM * sizeof(bf16_t);
  bf16_t* seq1b  = (bf16_t*)take(seq_bytes);
  bf16_t* seq2b  = (bf16_t*)take(seq_bytes);
  bf16_t* p1b    = (bf16_t*)take(seq_bytes);
  bf16_t* p2b    = (bf16_t*)take(seq_bytes);
  bf16_t* innerb = (bf16_t*)take(seq_bytes);
  bf16_t* w1b    = (bf16_t*)take(w_bytes);
  bf16_t* w2b    = (bf16_t*)take(w_bytes);
  bf16_t* wib    = (bf16_t*)take(w_bytes);
  float* ctx1    = (float*)take(KDIM * sizeof(float));
  float* ctx2    = (float*)take(KDIM * sizeof(float));
  float* p1sum   = (float*)take(KDIM * sizeof(float));
  float* sb      = (float*)take(KDIM * sizeof(float));
  float* logitsA = (float*)take(NROWS * sizeof(float));
  float* logitsB = (float*)take(NROWS * sizeof(float));
  float* wA      = (float*)take(NROWS * sizeof(float));
  float* wB      = (float*)take(NROWS * sizeof(float));

  // 1) zero f32 accumulators + output
  zero_kernel<<<(KDIM + 255) / 256, 256, 0, stream>>>(p1sum, KDIM);
  zero_kernel<<<(KDIM + 255) / 256, 256, 0, stream>>>(sb, KDIM);
  zero_kernel<<<(2 * KDIM + 255) / 256, 256, 0, stream>>>(out, 2 * KDIM);

  // 2) convert to bf16
  const int seq_n4 = NROWS * KDIM / 4;
  const int w_n4   = KDIM * KDIM / 4;
  cvt_kernel<<<(seq_n4 + 255) / 256, 256, 0, stream>>>((const float4*)seq1,  (v4bf*)seq1b, seq_n4);
  cvt_kernel<<<(seq_n4 + 255) / 256, 256, 0, stream>>>((const float4*)seq2,  (v4bf*)seq2b, seq_n4);
  cvt_kernel<<<(w_n4 + 255) / 256, 256, 0, stream>>>((const float4*)Wmlp1,  (v4bf*)w1b, w_n4);
  cvt_kernel<<<(w_n4 + 255) / 256, 256, 0, stream>>>((const float4*)Wmlp2,  (v4bf*)w2b, w_n4);
  cvt_kernel<<<(w_n4 + 255) / 256, 256, 0, stream>>>((const float4*)Winner, (v4bf*)wib, w_n4);

  // 3) ctx projections
  ctx_gemv_kernel<<<2 * KDIM, 256, 0, stream>>>(Wctx1, Wctx2, context, ctx1, ctx2);

  // 4) the three TDM-staged WMMA GEMMs (fused ELU + column sums)
  dim3 ggrid(NROWS / BM, KDIM / BN);
  gemm_bf16_kernel<<<ggrid, 256, 0, stream>>>(seq1b, w1b, bmlp1, ctx1, p1b, p1sum, 1);
  gemm_bf16_kernel<<<ggrid, 256, 0, stream>>>(seq2b, w2b, bmlp2, ctx2, p2b, nullptr, 1);
  gemm_bf16_kernel<<<ggrid, 256, 0, stream>>>(p2b, wib, nullptr, nullptr, innerb, sb, 0);

  // 5) logits
  rowdot_kernel<<<NROWS, 256, 0, stream>>>(p1b, sb, logitsA);
  rowdot_kernel<<<NROWS, 256, 0, stream>>>(innerb, p1sum, logitsB);

  // 6) softmax weights
  softmax_kernel<<<1, 256, 0, stream>>>(logitsA, wA, NROWS);
  softmax_kernel<<<1, 256, 0, stream>>>(logitsB, wB, NROWS);

  // 7) attention outputs (f32 sequences for accuracy)
  dim3 agrid(KDIM / 256, NROWS / 256);
  att_kernel<<<agrid, 256, 0, stream>>>(seq1, wA, out);
  att_kernel<<<agrid, 256, 0, stream>>>(seq2, wB, out + KDIM);
}
